// DyHGCN_H_74148315398747
// MI455X (gfx1250) — matrix operand order repaired and checked by reading, not compiled, not run on hardware
//
#include <hip/hip_runtime.h>
#include <math.h>

// ---------------- problem constants (from reference) ----------------
#define BB   16      // batch
#define L1C  201
#define LC   200     // sequence length
#define NN   30000   // nodes / vocab
#define DD   64      // node embedding dim
#define TT   8       // timesteps
#define EE   480000  // edges per timestep
#define HH   8       // heads
#define DKC  64      // head dim
#define INC  72      // D + 8
#define NBLK 40      // LC / STEP

typedef __attribute__((ext_vector_type(16))) __bf16        bf16x16;
typedef __attribute__((ext_vector_type(8)))  float         f32x8;
typedef __attribute__((ext_vector_type(4)))  unsigned int  u32x4;

// ---------------- elementwise / graph kernels ----------------

__global__ void fill_kernel(float* p, float v, int n) {
  int i = blockIdx.x * blockDim.x + threadIdx.x;
  if (i < n) p[i] = v;
}

__global__ void deg_accum_kernel(float* deg, const int* __restrict__ dst, int e) {
  int i = blockIdx.x * blockDim.x + threadIdx.x;
  if (i < e) atomicAdd(&deg[dst[i]], 1.0f);
}

__global__ void rsqrt_kernel(float* nrm, const float* __restrict__ deg, int n) {
  int i = blockIdx.x * blockDim.x + threadIdx.x;
  if (i < n) nrm[i] = rsqrtf(deg[i]);
}

// out[dst] += h[src] * (norm[src]*norm[dst]) for every edge, per feature dim
__global__ void scatter_kernel(float* out, const float* __restrict__ h,
                               const int* __restrict__ src, const int* __restrict__ dst,
                               const float* __restrict__ nrm, int e, int dim) {
  int i = blockIdx.x * blockDim.x + threadIdx.x;
  if (i >= e * dim) return;
  int ei = i / dim, d = i - ei * dim;
  int s = src[ei], t = dst[ei];
  atomicAdd(&out[t * dim + d], h[s * dim + d] * (nrm[s] * nrm[t]));
}

// out = scat + h/deg + bias
__global__ void combine_kernel(float* out, const float* __restrict__ scat,
                               const float* __restrict__ h, const float* __restrict__ deg,
                               const float* __restrict__ bias, int n, int dim) {
  int i = blockIdx.x * blockDim.x + threadIdx.x;
  if (i >= n * dim) return;
  int node = i / dim, d = i - node * dim;
  out[i] = scat[i] + h[i] / deg[node] + bias[d];
}

// per-block (over B and STEP) max of timestamps -> clipped history index
__global__ void his_kernel(const int* __restrict__ ts, int* his) {
  int j = threadIdx.x;
  if (j >= NBLK) return;
  int m = 0;
  for (int b = 0; b < BB; ++b)
    for (int s = 0; s < 5; ++s) {
      int v = ts[b * L1C + j * 5 + s];
      m = v > m ? v : m;
    }
  int h = m - 1;
  h = h < 0 ? 0 : (h > TT - 1 ? TT - 1 : h);
  his[j] = h;
}

// x[b,l,:] = concat(dyn[his[l/5], inp[b,l], :], pos_emb[l,:])
__global__ void build_x_kernel(float* x, const float* __restrict__ dyn,
                               const int* __restrict__ input, const int* __restrict__ his,
                               const float* __restrict__ pos) {
  int i = blockIdx.x * blockDim.x + threadIdx.x;
  if (i >= BB * LC * INC) return;
  int d = i % INC;
  int bl = i / INC;
  int l = bl % LC, b = bl / LC;
  if (d < DD) {
    int node = input[b * L1C + l];
    int hp = his[l / 5];
    x[i] = dyn[((long long)hp * NN + node) * DD + d];
  } else {
    x[i] = pos[l * 8 + (d - DD)];
  }
}

// causal + query-pad mask (replace with NEG_ATTN) then softmax, in place
__global__ void mask_softmax_kernel(float* sc, const int* __restrict__ input) {
  int i = blockIdx.x * blockDim.x + threadIdx.x;
  if (i >= BB * HH * LC) return;
  int q = i % LC;
  int bh = i / LC;
  int b = bh / HH;
  float* row = sc + ((long long)bh * LC + q) * LC;
  bool qpad = (input[b * L1C + q] == 0);
  const float NEGA = -4294967295.0f;   // -2^32 + 1
  float mx = -3.4e38f;
  for (int k = 0; k < LC; ++k) {
    float s = (k > q || qpad) ? NEGA : row[k];
    row[k] = s;
    mx = fmaxf(mx, s);
  }
  float sum = 0.0f;
  for (int k = 0; k < LC; ++k) {
    float e = __expf(row[k] - mx);
    row[k] = e;
    sum += e;
  }
  float inv = 1.0f / sum;
  for (int k = 0; k < LC; ++k) row[k] *= inv;
}

// out = LayerNorm(a + b) * g + beta   (row length INC)
__global__ void add_ln_kernel(float* out, const float* __restrict__ a,
                              const float* __restrict__ b,
                              const float* __restrict__ g, const float* __restrict__ beta) {
  int r = blockIdx.x * blockDim.x + threadIdx.x;
  if (r >= BB * LC) return;
  const float* pa = a + (long long)r * INC;
  const float* pb = b + (long long)r * INC;
  float mu = 0.0f;
  for (int d = 0; d < INC; ++d) mu += pa[d] + pb[d];
  mu *= (1.0f / INC);
  float var = 0.0f;
  for (int d = 0; d < INC; ++d) {
    float t = pa[d] + pb[d] - mu;
    var += t * t;
  }
  var *= (1.0f / INC);
  float is = rsqrtf(var + 1e-5f);
  float* po = out + (long long)r * INC;
  for (int d = 0; d < INC; ++d)
    po[d] = (pa[d] + pb[d] - mu) * is * g[d] + beta[d];
}

// set -inf at previously-seen items (and column 0 from the tril-zero fill)
__global__ void prev_mask_kernel(float* out, const int* __restrict__ input) {
  int i = blockIdx.x * blockDim.x + threadIdx.x;
  if (i >= BB * LC) return;
  int q = i % LC, b = i / LC;
  float* row = out + (long long)i * NN;
  const float NI = -__builtin_inff();
  if (q < LC - 1) row[0] = NI;
  const int* inpb = input + b * L1C;
  for (int k = 0; k <= q; ++k) row[inpb[k]] = NI;
}

// ---------------- generic batched WMMA GEMM (bf16 in, f32 acc) ----------------
// C[bb] = alpha * A[bb] @ op(B[bb]) + bias, optional ReLU.
// Block tile 64(M) x 64(N), BK=32; 8 waves, each wave computes a 16x32 strip:
// one A fragment reused across two B fragments -> 2 WMMAs per 6 ds_load_b128.
// Double-buffered LDS + register-staged pipeline (next tile's global loads in
// flight during the WMMAs). A row-major / B transposed in LDS, row stride
// 40 bf16 -> 16B aligned, conflict-free bank rotation (gcd(20,64)=4).
#define AST 40
#define BST 40

__global__ __launch_bounds__(256) void wmma_gemm_kernel(
    const float* __restrict__ A, const float* __restrict__ Bm, float* __restrict__ C,
    const float* __restrict__ bias,
    int M, int N, int K, int lda, int ldb, int ldc,
    int transB, int relu, float alpha,
    long long sAo, long long sAi, long long sBo, long long sBi,
    long long sCo, long long sCi, int innerB)
{
  __shared__ alignas(16) __bf16 As[2][64 * AST];
  __shared__ alignas(16) __bf16 BsT[2][64 * BST];

  const int tid  = threadIdx.x;
  const int wave = tid >> 5;
  const int lane = tid & 31;
  const int hv   = (lane >> 4) & 1;
  const int l15  = lane & 15;
  const int wm   = (wave & 3) << 4;   // 4 row groups of 16
  const int wn   = (wave >> 2) << 5;  // 2 col groups of 32
  const int bn0  = blockIdx.x * 64;
  const int bm0  = blockIdx.y * 64;

  const int bb = blockIdx.z;
  const int ob = bb / innerB, ib = bb - ob * innerB;
  const long long Aoff = (long long)ob * sAo + (long long)ib * sAi;
  const long long Boff = (long long)ob * sBo + (long long)ib * sBi;
  const long long Coff = (long long)ob * sCo + (long long)ib * sCi;

  // staging thread mapping
  const int aR  = tid >> 2, aK  = (tid & 3) << 3;   // A: row 0..63, 8 consecutive k
  const int bKr = tid >> 3, bNc = (tid & 7) << 3;   // B normal: 8 consecutive n
  const int bNr = tid >> 2, bKc = (tid & 3) << 3;   // B trans : 8 consecutive k

  float ar[8];
  float br[8];

  auto loadA = [&](int k0) {
    int gr = bm0 + aR;
    const float* base = A + Aoff;
    if (bm0 + 64 <= M && k0 + 32 <= K) {
      const float* p = base + (long long)gr * lda + (k0 + aK);
      float4 t0 = *(const float4*)p;
      float4 t1 = *(const float4*)(p + 4);
      ar[0] = t0.x; ar[1] = t0.y; ar[2] = t0.z; ar[3] = t0.w;
      ar[4] = t1.x; ar[5] = t1.y; ar[6] = t1.z; ar[7] = t1.w;
      if (k0 + 64 <= K) __builtin_prefetch(p + 32, 0, 1);
    } else {
      int cr = gr < M ? gr : M - 1;               // clamp -> unconditional loads
#pragma unroll
      for (int u = 0; u < 8; ++u) {
        int gk = k0 + aK + u;
        int ck = gk < K ? gk : K - 1;
        float v = base[(long long)cr * lda + ck];
        ar[u] = (gr < M && gk < K) ? v : 0.0f;
      }
    }
  };

  auto loadB = [&](int k0) {
    const float* base = Bm + Boff;
    if (!transB) {
      int gk = k0 + bKr;
      if (k0 + 32 <= K && bn0 + 64 <= N) {
        const float* p = base + (long long)gk * ldb + (bn0 + bNc);
        float4 t0 = *(const float4*)p;
        float4 t1 = *(const float4*)(p + 4);
        br[0] = t0.x; br[1] = t0.y; br[2] = t0.z; br[3] = t0.w;
        br[4] = t1.x; br[5] = t1.y; br[6] = t1.z; br[7] = t1.w;
      } else {
        int ck = gk < K ? gk : K - 1;
#pragma unroll
        for (int u = 0; u < 8; ++u) {
          int gn = bn0 + bNc + u;
          int cn = gn < N ? gn : N - 1;
          float v = base[(long long)ck * ldb + cn];
          br[u] = (gk < K && gn < N) ? v : 0.0f;
        }
      }
    } else {
      int gn = bn0 + bNr;
      if (bn0 + 64 <= N && k0 + 32 <= K) {
        const float* p = base + (long long)gn * ldb + (k0 + bKc);
        float4 t0 = *(const float4*)p;
        float4 t1 = *(const float4*)(p + 4);
        br[0] = t0.x; br[1] = t0.y; br[2] = t0.z; br[3] = t0.w;
        br[4] = t1.x; br[5] = t1.y; br[6] = t1.z; br[7] = t1.w;
      } else {
        int cn = gn < N ? gn : N - 1;
#pragma unroll
        for (int u = 0; u < 8; ++u) {
          int gk = k0 + bKc + u;
          int ck = gk < K ? gk : K - 1;
          float v = base[(long long)cn * ldb + ck];
          br[u] = (gn < N && gk < K) ? v : 0.0f;
        }
      }
    }
  };

  auto storeTile = [&](int buf) {
    // A: 8 bf16 contiguous -> one b128 store
    {
      union { u32x4 q; __bf16 h[8]; } p;
#pragma unroll
      for (int u = 0; u < 8; ++u) p.h[u] = (__bf16)ar[u];
      *(u32x4*)&As[buf][aR * AST + aK] = p.q;
    }
    if (!transB) {
      // transpose into BsT: 8 scattered b16 stores
#pragma unroll
      for (int u = 0; u < 8; ++u)
        BsT[buf][(bNc + u) * BST + bKr] = (__bf16)br[u];
    } else {
      // already k-contiguous: one b128 store
      union { u32x4 q; __bf16 h[8]; } p;
#pragma unroll
      for (int u = 0; u < 8; ++u) p.h[u] = (__bf16)br[u];
      *(u32x4*)&BsT[buf][bNr * BST + bKc] = p.q;
    }
  };

  f32x8 acc0 = {};
  f32x8 acc1 = {};
  loadA(0);
  loadB(0);
  int buf = 0;

  for (int k0 = 0; k0 < K; k0 += 32) {
    storeTile(buf);
    __syncthreads();
    if (k0 + 32 < K) { loadA(k0 + 32); loadB(k0 + 32); }   // next tile in flight

    // A fragment: 16 contiguous k values (k = hv*16..hv*16+15)
    const __bf16* apL  = &As[buf][(wm + l15) * AST + (hv << 4)];
    // B fragments: 16 contiguous k values for columns wn+l15 and wn+16+l15
    const __bf16* bpL0 = &BsT[buf][(wn + l15) * BST + (hv << 4)];
    const __bf16* bpL1 = &BsT[buf][(wn + 16 + l15) * BST + (hv << 4)];

    union { u32x4 q[2]; bf16x16 v; } fa, fb0, fb1;
    fa.q[0]  = *(const u32x4*)apL;
    fa.q[1]  = *(const u32x4*)(apL + 8);
    fb0.q[0] = *(const u32x4*)bpL0;
    fb0.q[1] = *(const u32x4*)(bpL0 + 8);
    fb1.q[0] = *(const u32x4*)bpL1;
    fb1.q[1] = *(const u32x4*)(bpL1 + 8);

    acc0 = __builtin_amdgcn_wmma_f32_16x16x32_bf16(
        false, fa.v, false, fb0.v, (short)0, acc0, false, false);
    acc1 = __builtin_amdgcn_wmma_f32_16x16x32_bf16(
        false, fa.v, false, fb1.v, (short)0, acc1, false, false);
    buf ^= 1;
  }

  // epilogue: lanes 0-15 -> M=r, lanes 16-31 -> M=r+8
  int r0  = bm0 + wm + (hv << 3);
  int cn0 = bn0 + wn + l15;
  int cn1 = cn0 + 16;
  float bv0 = (bias != nullptr && cn0 < N) ? bias[cn0] : 0.0f;
  float bv1 = (bias != nullptr && cn1 < N) ? bias[cn1] : 0.0f;
#pragma unroll
  for (int r = 0; r < 8; ++r) {
    int row = r0 + r;
    if (row < M) {
      if (cn0 < N) {
        float v = acc0[r] * alpha + bv0;
        if (relu) v = fmaxf(v, 0.0f);
        C[Coff + (long long)row * ldc + cn0] = v;
      }
      if (cn1 < N) {
        float v = acc1[r] * alpha + bv1;
        if (relu) v = fmaxf(v, 0.0f);
        C[Coff + (long long)row * ldc + cn1] = v;
      }
    }
  }
}

// ---------------- host side ----------------

static inline dim3 g1(int n) { return dim3((n + 255) / 256); }

static void launch_gemm(hipStream_t st, const float* A, const float* Bm, float* C,
                        const float* bias, int M, int N, int K,
                        int lda, int ldb, int ldc, bool transB, bool relu, float alpha,
                        int batches = 1, int innerB = 1,
                        long long sAo = 0, long long sAi = 0,
                        long long sBo = 0, long long sBi = 0,
                        long long sCo = 0, long long sCi = 0)
{
  dim3 grid((N + 63) / 64, (M + 63) / 64, batches);
  wmma_gemm_kernel<<<grid, dim3(256), 0, st>>>(
      A, Bm, C, bias, M, N, K, lda, ldb, ldc,
      transB ? 1 : 0, relu ? 1 : 0, alpha,
      sAo, sAi, sBo, sBi, sCo, sCi, innerB);
}

extern "C" void kernel_launch(void* const* d_in, const int* in_sizes, int n_in,
                              void* d_out, int out_size, void* d_ws, size_t ws_size,
                              hipStream_t stream)
{
  const int*   input = (const int*)d_in[0];
  const int*   its   = (const int*)d_in[1];
  const int*   eidx  = (const int*)d_in[2];
  const float* emb_g = (const float*)d_in[3];
  const float* g1w   = (const float*)d_in[4];
  const float* g1b   = (const float*)d_in[5];
  const float* g2w   = (const float*)d_in[6];
  const float* g2b   = (const float*)d_in[7];
  const float* pos   = (const float*)d_in[8];
  const float* Wq    = (const float*)d_in[9];
  const float* Wk    = (const float*)d_in[10];
  const float* Wv    = (const float*)d_in[11];
  const float* Wo    = (const float*)d_in[12];
  const float* f1w   = (const float*)d_in[13];
  const float* f1b   = (const float*)d_in[14];
  const float* f2w   = (const float*)d_in[15];
  const float* f2b   = (const float*)d_in[16];
  const float* lng   = (const float*)d_in[17];
  const float* lnb   = (const float*)d_in[18];
  const float* ow    = (const float*)d_in[19];
  const float* obv   = (const float*)d_in[20];
  float* out = (float*)d_out;

  // ---- workspace layout (~114 MB of f32) ----
  float* ws  = (float*)d_ws;
  float* dyn = ws;              ws += (size_t)TT * NN * DD;    // persists until build_x
  int*   his = (int*)ws;        ws += 64;
  float* x   = ws;              ws += (size_t)BB * LC * INC;   // persists to the end
  float* scratch = ws;                                         // GCN/attn regions overlap

  // GCN-phase aliases
  float* deg = scratch;
  float* nrm = deg + NN;
  float* h1  = nrm + NN;
  float* s1  = h1 + (size_t)NN * 128;
  float* h2  = s1 + (size_t)NN * 128;
  float* s2  = h2 + (size_t)NN * DD;

  // ---- dynamic graph: 2-layer GCN per timestep ----
  for (int t = 0; t < TT; ++t) {
    const int* src = eidx + (size_t)t * 2 * EE;
    const int* dst = src + EE;
    fill_kernel<<<g1(NN), 256, 0, stream>>>(deg, 1.0f, NN);
    deg_accum_kernel<<<g1(EE), 256, 0, stream>>>(deg, dst, EE);
    rsqrt_kernel<<<g1(NN), 256, 0, stream>>>(nrm, deg, NN);

    // layer 1: h1 = emb_g @ gcn1_w   (30000x64 @ 64x128)
    launch_gemm(stream, emb_g, g1w, h1, nullptr, NN, 128, 64, 64, 128, 128, false, false, 1.0f);
    fill_kernel<<<g1(NN * 128), 256, 0, stream>>>(s1, 0.0f, NN * 128);
    scatter_kernel<<<g1(EE * 128), 256, 0, stream>>>(s1, h1, src, dst, nrm, EE, 128);
    combine_kernel<<<g1(NN * 128), 256, 0, stream>>>(s1, s1, h1, deg, g1b, NN, 128);

    // layer 2: h2 = s1 @ gcn2_w      (30000x128 @ 128x64)
    launch_gemm(stream, s1, g2w, h2, nullptr, NN, 64, 128, 128, 64, 64, false, false, 1.0f);
    fill_kernel<<<g1(NN * 64), 256, 0, stream>>>(s2, 0.0f, NN * 64);
    scatter_kernel<<<g1(EE * 64), 256, 0, stream>>>(s2, h2, src, dst, nrm, EE, 64);
    combine_kernel<<<g1(NN * 64), 256, 0, stream>>>(dyn + (size_t)t * NN * DD,
                                                    s2, h2, deg, g2b, NN, 64);
  }

  // ---- gather dynamic embeddings + positional features ----
  his_kernel<<<1, 64, 0, stream>>>(its, his);
  build_x_kernel<<<g1(BB * LC * INC), 256, 0, stream>>>(x, dyn, input, his, pos);

  // attention-phase aliases (GCN scratch is dead now)
  float* q   = scratch;
  float* kx  = q   + (size_t)BB * LC * 512;
  float* v   = kx  + (size_t)BB * LC * 512;
  float* sc  = v   + (size_t)BB * LC * 512;
  float* vat = sc  + (size_t)BB * HH * LC * LC;
  float* ao  = vat + (size_t)BB * LC * 512;
  float* x1  = ao  + (size_t)BB * LC * INC;
  float* fh  = x1  + (size_t)BB * LC * INC;
  float* fo  = fh  + (size_t)BB * LC * INC;
  float* x2  = fo  + (size_t)BB * LC * INC;

  const int BL = BB * LC;
  // q/k/v projections: (3200x72) @ (72x512)
  launch_gemm(stream, x, Wq, q,  nullptr, BL, 512, INC, INC, 512, 512, false, false, 1.0f);
  launch_gemm(stream, x, Wk, kx, nullptr, BL, 512, INC, INC, 512, 512, false, false, 1.0f);
  launch_gemm(stream, x, Wv, v,  nullptr, BL, 512, INC, INC, 512, 512, false, false, 1.0f);

  // scores[b,h] = alpha * q[b,:,h,:] @ k[b,:,h,:]^T   (128 batched 200x200x64, NT)
  float alpha = 1.0f / (sqrtf((float)DKC) + 1e-6f);
  launch_gemm(stream, q, kx, sc, nullptr, LC, LC, DKC, 512, 512, LC, true, false, alpha,
              BB * HH, HH,
              (long long)LC * 512, 64,
              (long long)LC * 512, 64,
              (long long)HH * LC * LC, (long long)LC * LC);

  mask_softmax_kernel<<<g1(BB * HH * LC), 256, 0, stream>>>(sc, input);

  // vat[b,:,h,:] = attn[b,h] @ v[b,:,h,:]   (128 batched 200x64x200, NN)
  launch_gemm(stream, sc, v, vat, nullptr, LC, DKC, LC, LC, 512, 512, false, false, 1.0f,
              BB * HH, HH,
              (long long)HH * LC * LC, (long long)LC * LC,
              (long long)LC * 512, 64,
              (long long)LC * 512, 64);

  // output projection + FFN + layernorms
  launch_gemm(stream, vat, Wo, ao, nullptr, BL, INC, 512, 512, INC, INC, false, false, 1.0f);
  add_ln_kernel<<<g1(BL), 256, 0, stream>>>(x1, x, ao, lng, lnb);
  launch_gemm(stream, x1, f1w, fh, f1b, BL, INC, INC, INC, INC, INC, false, true,  1.0f);
  launch_gemm(stream, fh, f2w, fo, f2b, BL, INC, INC, INC, INC, INC, false, false, 1.0f);
  add_ln_kernel<<<g1(BL), 256, 0, stream>>>(x2, x1, fo, lng, lnb);

  // logits = x2 @ out_w^T + out_b   ((3200x72) @ (72x30000), B transposed)
  launch_gemm(stream, x2, ow, out, obv, BL, NN, INC, INC, INC, NN, true, false, 1.0f);
  prev_mask_kernel<<<g1(BL), 256, 0, stream>>>(out, input);

  (void)in_sizes; (void)n_in; (void)out_size; (void)ws_size;
}
